// SLinOSSMixer_90074054132527
// MI455X (gfx1250) — matrix operand order repaired and checked by reading, not compiled, not run on hardware
//
#include <hip/hip_runtime.h>
#include <hip/hip_bf16.h>

#define BATCH   2
#define SEQ     2048
#define BT      (BATCH * SEQ)                       // 4096
#define DMODEL  512
#define DINNER  1024
#define NHEADS  16
#define DHEAD   64
#define DSTATE  32
#define PROJ    (2 * DINNER + NHEADS + NHEADS * 4 * DSTATE)  // 4112
#define CH      64                                  // scan chunk length
#define NCHUNK  (SEQ / CH)                          // 32

typedef __attribute__((ext_vector_type(16))) __bf16        v16bf;
typedef __attribute__((ext_vector_type(8)))  float         v8f;
typedef __attribute__((ext_vector_type(8)))  unsigned int  v8u;

// Round-to-nearest-even f32 -> bf16, packed pair into one dword.
__device__ __forceinline__ unsigned int pack_bf16x2(float lo, float hi) {
  unsigned int a = __float_as_uint(lo);
  unsigned int b = __float_as_uint(hi);
  a += 0x7FFFu + ((a >> 16) & 1u);
  b += 0x7FFFu + ((b >> 16) & 1u);
  return (a >> 16) | (b & 0xFFFF0000u);
}

__device__ __forceinline__ float silu_f(float x) {
  return x * (1.0f / (1.0f + __expf(-x)));
}

// One-time pack: f32 array -> bf16x2 dwords.
__global__ __launch_bounds__(256)
void pack_bf16_kernel(const float* __restrict__ src, unsigned int* __restrict__ dst,
                      int n_pairs) {
  const int i = blockIdx.x * 256 + threadIdx.x;
  if (i < n_pairs) {
    const float2 f = *(const float2*)(src + 2 * (size_t)i);
    dst[i] = pack_bf16x2(f.x, f.y);
  }
}

// D(MxN,f32) = A(MxK) * B(NxK)^T, A/B pre-packed bf16 (dword = 2 consecutive K).
// Each wave computes a 32(M) x 64(N) strip: 2 A frags x 4 B frags -> 8 WMMAs / K-chunk.
__global__ __launch_bounds__(256)
void wmma_gemm_bf16_kernel(const unsigned int* __restrict__ A,
                           const unsigned int* __restrict__ Bw,
                           float* __restrict__ D, int M, int N, int K) {
  const int lane = threadIdx.x & 31;
  // Force wave-uniform tile indices into SGPRs -> scalar branches, EXEC untouched.
  const int wave = __builtin_amdgcn_readfirstlane(blockIdx.x * 8 + (threadIdx.x >> 5));
  const int Mtiles = M >> 5;                   // 32-row tiles
  const int Ntiles = N >> 4;
  const int ngroups = (Ntiles + 3) >> 2;
  const int mtile = wave / ngroups;
  const int ngrp  = wave - mtile * ngroups;
  if (mtile >= Mtiles) return;                 // scalar exit
  const int nrem = __builtin_amdgcn_readfirstlane(Ntiles - ngrp * 4);  // 1..4

  const int m    = lane & 15;
  const int half = lane >> 4;
  const int Kd   = K >> 1;                     // dwords per row
  const unsigned int* Arow0 = A + (size_t)(mtile * 32 + m) * Kd;
  const unsigned int* Arow1 = Arow0 + (size_t)16 * Kd;

  v8f acc[2][4] = {};

  for (int k0 = 0; k0 < Kd; k0 += 16) {        // 32 bf16 of K per chunk
    if (k0 + 16 < Kd) __builtin_prefetch(Arow0 + k0 + 16 + 4 * half, 0, 0);
    const uint4 a0lo = *(const uint4*)(Arow0 + k0 + 4 * half);
    const uint4 a0hi = *(const uint4*)(Arow0 + k0 + 8 + 4 * half);
    const uint4 a1lo = *(const uint4*)(Arow1 + k0 + 4 * half);
    const uint4 a1hi = *(const uint4*)(Arow1 + k0 + 8 + 4 * half);
    const v8u ap0 = {a0lo.x, a0lo.y, a0lo.z, a0lo.w, a0hi.x, a0hi.y, a0hi.z, a0hi.w};
    const v8u ap1 = {a1lo.x, a1lo.y, a1lo.z, a1lo.w, a1hi.x, a1hi.y, a1hi.z, a1hi.w};
    const v16bf a0 = __builtin_bit_cast(v16bf, ap0);
    const v16bf a1 = __builtin_bit_cast(v16bf, ap1);

#pragma unroll
    for (int j = 0; j < 4; ++j) {
      if (j < nrem) {                          // scalar condition
        const unsigned int* Brow = Bw + (size_t)((ngrp * 4 + j) * 16 + m) * Kd;
        const uint4 b0 = *(const uint4*)(Brow + k0 + 4 * half);
        const uint4 b1 = *(const uint4*)(Brow + k0 + 8 + 4 * half);
        const v8u bp = {b0.x, b0.y, b0.z, b0.w, b1.x, b1.y, b1.z, b1.w};
        const v16bf bv = __builtin_bit_cast(v16bf, bp);
        acc[0][j] = __builtin_amdgcn_wmma_f32_16x16x32_bf16(
            false, a0, false, bv, (short)0, acc[0][j], false, false);
        acc[1][j] = __builtin_amdgcn_wmma_f32_16x16x32_bf16(
            false, a1, false, bv, (short)0, acc[1][j], false, false);
      }
    }
  }

#pragma unroll
  for (int s = 0; s < 2; ++s) {
#pragma unroll
    for (int j = 0; j < 4; ++j) {
      if (j < nrem) {
        const int ncol = (ngrp * 4 + j) * 16 + m;
#pragma unroll
        for (int r = 0; r < 8; ++r) {
          D[(size_t)(mtile * 32 + s * 16 + r + 8 * half) * N + ncol] = acc[s][j][r];
        }
      }
    }
  }
}

// Causal 4-tap depthwise conv over time + bias + SiLU -> u (B,T,DINNER)
__global__ __launch_bounds__(256)
void conv_silu_kernel(const float* __restrict__ proj, const float* __restrict__ w,
                      const float* __restrict__ bias, float* __restrict__ u) {
  const int idx = blockIdx.x * 256 + threadIdx.x;      // over BT*DINNER
  if (idx >= BT * DINNER) return;
  const int c  = idx & (DINNER - 1);
  const int bt = idx >> 10;
  const int t  = bt & (SEQ - 1);
  float acc = bias[c];
#pragma unroll
  for (int j = 0; j < 4; ++j) {
    const int tj = t - 3 + j;
    if (tj >= 0) {
      const float v = proj[(size_t)(bt - 3 + j) * PROJ + DINNER + c];
      acc += v * w[c * 4 + j];
    }
  }
  u[idx] = silu_f(acc);
}

// dt = clip(softplus(params + dt_bias), 0.03, 0.1)
__global__ __launch_bounds__(256)
void dt_kernel(const float* __restrict__ proj, const float* __restrict__ dt_bias,
               float* __restrict__ dt) {
  const int idx = blockIdx.x * 256 + threadIdx.x;      // over BT*NHEADS
  if (idx >= BT * NHEADS) return;
  const int h  = idx & (NHEADS - 1);
  const int bt = idx >> 4;
  const float p = proj[(size_t)bt * PROJ + 2 * DINNER + h] + dt_bias[h];
  const float sp = (p > 20.0f) ? p : log1pf(__expf(p));
  dt[idx] = fminf(fmaxf(sp, 0.03f), 0.1f);
}

// Precompute lambda = exp(-dt*gamma) * (cos, sin)(dt*theta) per (b,t,h,n).
__global__ __launch_bounds__(256)
void lam_kernel(const float* __restrict__ dt, const float* __restrict__ gamma,
                const float* __restrict__ theta,
                float* __restrict__ lam_re, float* __restrict__ lam_im) {
  const int idx = blockIdx.x * 256 + threadIdx.x;      // over BT*NHEADS*DSTATE
  if (idx >= BT * NHEADS * DSTATE) return;
  const int n   = idx & (DSTATE - 1);
  const int h   = (idx >> 5) & (NHEADS - 1);
  const int bth = idx >> 5;                            // bt*NHEADS + h
  const float dtv = dt[bth];
  const float dec = __expf(-dtv * gamma[h * DSTATE + n]);
  const float ph  = dtv * theta[h * DSTATE + n];
  lam_re[idx] = dec * __cosf(ph);
  lam_im[idx] = dec * __sinf(ph);
}

// ---- Chunked parallel scan ----
// Pass A: per (b,h,chunk): local state S_c with zero init + cumulative lambda product P_c.
__global__ __launch_bounds__(64)
void scan_partial_kernel(const float* __restrict__ proj, const float* __restrict__ u,
                         const float* __restrict__ dt,
                         const float* __restrict__ lam_re, const float* __restrict__ lam_im,
                         float* __restrict__ sP_re, float* __restrict__ sP_im,
                         float* __restrict__ pP_re, float* __restrict__ pP_im) {
  const int c = blockIdx.x & (NCHUNK - 1);
  const int h = (blockIdx.x >> 5) & (NHEADS - 1);
  const int b = blockIdx.x >> 9;
  const int p = threadIdx.x;

  __shared__ float s_bre[DSTATE], s_bim[DSTATE], s_brep[DSTATE], s_bimp[DSTATE];
  __shared__ float s_lre[DSTATE], s_lim[DSTATE];
  __shared__ float s_hdt;

  float sre[DSTATE], sim[DSTATE];
#pragma unroll
  for (int n = 0; n < DSTATE; ++n) { sre[n] = 0.0f; sim[n] = 0.0f; }
  const size_t bcbase = 2 * DINNER + NHEADS + (size_t)h * (4 * DSTATE);
  const int t0 = c * CH;
  float u_prev = (t0 > 0) ? u[(size_t)(b * SEQ + t0 - 1) * DINNER + h * DHEAD + p] : 0.0f;
  if (p < DSTATE) {
    if (t0 > 0) {
      const float* pr = proj + (size_t)(b * SEQ + t0 - 1) * PROJ + bcbase;
      s_brep[p] = pr[p]; s_bimp[p] = pr[DSTATE + p];
    } else { s_brep[p] = 0.0f; s_bimp[p] = 0.0f; }
  }
  float ppr = 1.0f, ppi = 0.0f;                        // lambda product (thread p<32, n=p)

  for (int t = t0; t < t0 + CH; ++t) {
    const int bt = b * SEQ + t;
    if (p < DSTATE) {
      const float* pr = proj + (size_t)bt * PROJ + bcbase;
      s_bre[p] = pr[p]; s_bim[p] = pr[DSTATE + p];
      const int li = (bt * NHEADS + h) * DSTATE + p;
      const float lr = lam_re[li], lim = lam_im[li];
      s_lre[p] = lr; s_lim[p] = lim;
      if (p == 0) s_hdt = 0.5f * dt[bt * NHEADS + h];
      const float nr = lr * ppr - lim * ppi;
      const float ni = lr * ppi + lim * ppr;
      ppr = nr; ppi = ni;
    }
    __syncthreads();
    const float hdt = s_hdt;
    const float ut  = u[(size_t)bt * DINNER + h * DHEAD + p];
#pragma unroll
    for (int n = 0; n < DSTATE; ++n) {
      const float ire = hdt * (s_bre[n] * ut + s_brep[n] * u_prev);
      const float iim = hdt * (s_bim[n] * ut + s_bimp[n] * u_prev);
      const float nre = s_lre[n] * sre[n] - s_lim[n] * sim[n] + ire;
      const float nim = s_lre[n] * sim[n] + s_lim[n] * sre[n] + iim;
      sre[n] = nre; sim[n] = nim;
    }
    u_prev = ut;
    __syncthreads();
    if (p < DSTATE) { s_brep[p] = s_bre[p]; s_bimp[p] = s_bim[p]; }
  }

  const size_t base = (((size_t)(b * NHEADS + h) * NCHUNK + c) * DHEAD + p) * DSTATE;
#pragma unroll
  for (int n = 0; n < DSTATE; ++n) { sP_re[base + n] = sre[n]; sP_im[base + n] = sim[n]; }
  if (p < DSTATE) {
    const size_t pb = ((size_t)(b * NHEADS + h) * NCHUNK + c) * DSTATE + p;
    pP_re[pb] = ppr; pP_im[pb] = ppi;
  }
}

// Pass B: boundary recurrence over chunks; overwrite S_c with chunk-initial state.
__global__ __launch_bounds__(256)
void scan_combine_kernel(float* __restrict__ s_re, float* __restrict__ s_im,
                         const float* __restrict__ p_re, const float* __restrict__ p_im) {
  const int idx = blockIdx.x * 256 + threadIdx.x;      // over B*H*P*N = 65536
  if (idx >= BATCH * NHEADS * DHEAD * DSTATE) return;
  const int n  = idx & (DSTATE - 1);
  const int p  = (idx >> 5) & (DHEAD - 1);
  const int bh = idx >> 11;
  float cr = 0.0f, ci = 0.0f;
  for (int c = 0; c < NCHUNK; ++c) {
    const size_t sb = (((size_t)bh * NCHUNK + c) * DHEAD + p) * DSTATE + n;
    const size_t pb = ((size_t)bh * NCHUNK + c) * DSTATE + n;
    const float Sr = s_re[sb], Si = s_im[sb];
    const float Pr = p_re[pb], Pi = p_im[pb];
    const float nr = Pr * cr - Pi * ci + Sr;
    const float ni = Pr * ci + Pi * cr + Si;
    s_re[sb] = cr; s_im[sb] = ci;                      // init state for chunk c
    cr = nr; ci = ni;
  }
}

// Pass C: re-run each chunk from its exact initial state; emit y.
__global__ __launch_bounds__(64)
void scan_final_kernel(const float* __restrict__ proj, const float* __restrict__ u,
                       const float* __restrict__ dt,
                       const float* __restrict__ lam_re, const float* __restrict__ lam_im,
                       const float* __restrict__ s_re_in, const float* __restrict__ s_im_in,
                       const float* __restrict__ d_skip, float* __restrict__ y) {
  const int c = blockIdx.x & (NCHUNK - 1);
  const int h = (blockIdx.x >> 5) & (NHEADS - 1);
  const int b = blockIdx.x >> 9;
  const int p = threadIdx.x;

  __shared__ float s_bre[DSTATE], s_bim[DSTATE], s_cre[DSTATE], s_cim[DSTATE];
  __shared__ float s_brep[DSTATE], s_bimp[DSTATE];
  __shared__ float s_lre[DSTATE], s_lim[DSTATE];
  __shared__ float s_hdt;

  float sre[DSTATE], sim[DSTATE];
  const size_t base = (((size_t)(b * NHEADS + h) * NCHUNK + c) * DHEAD + p) * DSTATE;
#pragma unroll
  for (int n = 0; n < DSTATE; ++n) { sre[n] = s_re_in[base + n]; sim[n] = s_im_in[base + n]; }

  const size_t bcbase = 2 * DINNER + NHEADS + (size_t)h * (4 * DSTATE);
  const int t0 = c * CH;
  float u_prev = (t0 > 0) ? u[(size_t)(b * SEQ + t0 - 1) * DINNER + h * DHEAD + p] : 0.0f;
  if (p < DSTATE) {
    if (t0 > 0) {
      const float* pr = proj + (size_t)(b * SEQ + t0 - 1) * PROJ + bcbase;
      s_brep[p] = pr[p]; s_bimp[p] = pr[DSTATE + p];
    } else { s_brep[p] = 0.0f; s_bimp[p] = 0.0f; }
  }
  const float dskip = d_skip[h];

  for (int t = t0; t < t0 + CH; ++t) {
    const int bt = b * SEQ + t;
    if (p < DSTATE) {
      const float* pr = proj + (size_t)bt * PROJ + bcbase;
      s_bre[p] = pr[p];              s_bim[p] = pr[DSTATE + p];
      s_cre[p] = pr[2 * DSTATE + p]; s_cim[p] = pr[3 * DSTATE + p];
      const int li = (bt * NHEADS + h) * DSTATE + p;
      s_lre[p] = lam_re[li]; s_lim[p] = lam_im[li];
      if (p == 0) s_hdt = 0.5f * dt[bt * NHEADS + h];
    }
    __syncthreads();
    const float hdt = s_hdt;
    const float ut  = u[(size_t)bt * DINNER + h * DHEAD + p];
    float yacc = 0.0f;
#pragma unroll
    for (int n = 0; n < DSTATE; ++n) {
      const float ire = hdt * (s_bre[n] * ut + s_brep[n] * u_prev);
      const float iim = hdt * (s_bim[n] * ut + s_bimp[n] * u_prev);
      const float nre = s_lre[n] * sre[n] - s_lim[n] * sim[n] + ire;
      const float nim = s_lre[n] * sim[n] + s_lim[n] * sre[n] + iim;
      sre[n] = nre; sim[n] = nim;
      yacc += nre * s_cre[n] - nim * s_cim[n];
    }
    y[(size_t)bt * DINNER + h * DHEAD + p] = yacc + dskip * ut;
    u_prev = ut;
    __syncthreads();
    if (p < DSTATE) { s_brep[p] = s_bre[p]; s_bimp[p] = s_bim[p]; }
  }
}

// Fallback: fully sequential scan (used only if workspace is too small for chunked path).
__global__ __launch_bounds__(64)
void scan_seq_kernel(const float* __restrict__ proj, const float* __restrict__ u,
                     const float* __restrict__ dt, const float* __restrict__ gamma,
                     const float* __restrict__ theta, const float* __restrict__ d_skip,
                     float* __restrict__ y) {
  const int b = blockIdx.x >> 4;
  const int h = blockIdx.x & (NHEADS - 1);
  const int p = threadIdx.x;
  __shared__ float s_bre[DSTATE], s_bim[DSTATE], s_cre[DSTATE], s_cim[DSTATE];
  __shared__ float s_brep[DSTATE], s_bimp[DSTATE];
  __shared__ float s_lre[DSTATE], s_lim[DSTATE];
  __shared__ float s_dt;
  float sre[DSTATE], sim[DSTATE];
#pragma unroll
  for (int n = 0; n < DSTATE; ++n) { sre[n] = 0.0f; sim[n] = 0.0f; }
  if (p < DSTATE) { s_brep[p] = 0.0f; s_bimp[p] = 0.0f; }
  float u_prev = 0.0f;
  const float dskip = d_skip[h];
  const size_t bcbase = 2 * DINNER + NHEADS + (size_t)h * (4 * DSTATE);
  for (int t = 0; t < SEQ; ++t) {
    const int bt = b * SEQ + t;
    if (p < DSTATE) {
      const float* pr = proj + (size_t)bt * PROJ + bcbase;
      s_bre[p] = pr[p];              s_bim[p] = pr[DSTATE + p];
      s_cre[p] = pr[2 * DSTATE + p]; s_cim[p] = pr[3 * DSTATE + p];
      const float dtv = dt[bt * NHEADS + h];
      if (p == 0) s_dt = dtv;
      const float dec = __expf(-dtv * gamma[h * DSTATE + p]);
      const float ph  = dtv * theta[h * DSTATE + p];
      s_lre[p] = dec * __cosf(ph);
      s_lim[p] = dec * __sinf(ph);
    }
    __syncthreads();
    const float hdt = 0.5f * s_dt;
    const float ut  = u[(size_t)bt * DINNER + h * DHEAD + p];
    float yacc = 0.0f;
#pragma unroll
    for (int n = 0; n < DSTATE; ++n) {
      const float ire = hdt * (s_bre[n] * ut + s_brep[n] * u_prev);
      const float iim = hdt * (s_bim[n] * ut + s_bimp[n] * u_prev);
      const float nre = s_lre[n] * sre[n] - s_lim[n] * sim[n] + ire;
      const float nim = s_lre[n] * sim[n] + s_lim[n] * sre[n] + iim;
      sre[n] = nre; sim[n] = nim;
      yacc += nre * s_cre[n] - nim * s_cim[n];
    }
    y[(size_t)bt * DINNER + h * DHEAD + p] = yacc + dskip * ut;
    u_prev = ut;
    __syncthreads();
    if (p < DSTATE) { s_brep[p] = s_bre[p]; s_bimp[p] = s_bim[p]; }
  }
}

// z = y * silu(gate); RMS-normalize over DINNER; emit packed bf16 z for output GEMM.
__global__ __launch_bounds__(256)
void gate_norm_kernel(const float* __restrict__ proj, const float* __restrict__ nw,
                      const float* __restrict__ y, unsigned int* __restrict__ zbf) {
  const int bt = blockIdx.x;
  const float* g  = proj + (size_t)bt * PROJ;
  const float* yr = y + (size_t)bt * DINNER;
  const int c0 = threadIdx.x * 4;
  float local[4];
  float ss = 0.0f;
#pragma unroll
  for (int i = 0; i < 4; ++i) {
    const int c = c0 + i;
    const float z = yr[c] * silu_f(g[c]);
    local[i] = z;
    ss += z * z;
  }
  __shared__ float red[256];
  red[threadIdx.x] = ss;
  __syncthreads();
  for (int s2 = 128; s2 > 0; s2 >>= 1) {
    if (threadIdx.x < s2) red[threadIdx.x] += red[threadIdx.x + s2];
    __syncthreads();
  }
  const float scale = rsqrtf(red[0] * (1.0f / DINNER) + 1e-5f);
  unsigned int* zr = zbf + (size_t)bt * (DINNER / 2);
  zr[c0 / 2]     = pack_bf16x2(local[0] * scale * nw[c0],
                               local[1] * scale * nw[c0 + 1]);
  zr[c0 / 2 + 1] = pack_bf16x2(local[2] * scale * nw[c0 + 2],
                               local[3] * scale * nw[c0 + 3]);
}

extern "C" void kernel_launch(void* const* d_in, const int* in_sizes, int n_in,
                              void* d_out, int out_size, void* d_ws, size_t ws_size,
                              hipStream_t stream) {
  const float* x       = (const float*)d_in[0];
  const float* W_in    = (const float*)d_in[1];
  const float* dw_w    = (const float*)d_in[2];
  const float* dw_b    = (const float*)d_in[3];
  const float* dt_bias = (const float*)d_in[4];
  const float* gamma   = (const float*)d_in[5];
  const float* theta   = (const float*)d_in[6];
  const float* d_skip  = (const float*)d_in[7];
  const float* nw      = (const float*)d_in[8];
  const float* W_out   = (const float*)d_in[9];
  float* out = (float*)d_out;

  float* ws   = (float*)d_ws;
  size_t off = 0;
  float* proj = ws + off;  off += (size_t)BT * PROJ;
  float* u    = ws + off;  off += (size_t)BT * DINNER;
  float* dtv  = ws + off;  off += (size_t)BT * NHEADS;
  float* y    = ws + off;  off += (size_t)BT * DINNER;
  unsigned int* xbf  = (unsigned int*)(ws + off);  off += (size_t)BT * DMODEL / 2;
  unsigned int* wibf = (unsigned int*)(ws + off);  off += (size_t)PROJ * DMODEL / 2;
  unsigned int* wobf = (unsigned int*)(ws + off);  off += (size_t)DMODEL * DINNER / 2;
  unsigned int* zbf  = (unsigned int*)(ws + off);  off += (size_t)BT * DINNER / 2;
  const size_t base_units = off;
  float* lam_re = ws + off;  off += (size_t)BT * NHEADS * DSTATE;
  float* lam_im = ws + off;  off += (size_t)BT * NHEADS * DSTATE;
  float* sP_re  = ws + off;  off += (size_t)BATCH * NHEADS * NCHUNK * DHEAD * DSTATE;
  float* sP_im  = ws + off;  off += (size_t)BATCH * NHEADS * NCHUNK * DHEAD * DSTATE;
  float* pP_re  = ws + off;  off += (size_t)BATCH * NHEADS * NCHUNK * DSTATE;
  float* pP_im  = ws + off;  off += (size_t)BATCH * NHEADS * NCHUNK * DSTATE;
  const bool chunked = ws_size >= off * sizeof(float);
  (void)base_units;

  // 0) One-time f32 -> bf16 packing of GEMM operands
  {
    const int np_x  = BT * DMODEL / 2;
    const int np_wi = PROJ * DMODEL / 2;
    const int np_wo = DMODEL * DINNER / 2;
    pack_bf16_kernel<<<(np_x  + 255) / 256, 256, 0, stream>>>(x,     xbf,  np_x);
    pack_bf16_kernel<<<(np_wi + 255) / 256, 256, 0, stream>>>(W_in,  wibf, np_wi);
    pack_bf16_kernel<<<(np_wo + 255) / 256, 256, 0, stream>>>(W_out, wobf, np_wo);
  }
  // 1) Input projection GEMM: proj = x @ W_in^T   (4096 x 4112, K=512)
  {
    const int Ntiles = PROJ / 16, ngroups = (Ntiles + 3) / 4;
    const int waves = (BT / 32) * ngroups;
    wmma_gemm_bf16_kernel<<<(waves + 7) / 8, 256, 0, stream>>>(xbf, wibf, proj,
                                                               BT, PROJ, DMODEL);
  }
  // 2) Depthwise conv + SiLU, and dt
  conv_silu_kernel<<<(BT * DINNER) / 256, 256, 0, stream>>>(proj, dw_w, dw_b, u);
  dt_kernel<<<(BT * NHEADS) / 256, 256, 0, stream>>>(proj, dt_bias, dtv);
  // 3) Scan: chunked parallel path (2048-step recurrence -> depth ~160), else fallback
  if (chunked) {
    lam_kernel<<<(BT * NHEADS * DSTATE) / 256, 256, 0, stream>>>(dtv, gamma, theta,
                                                                 lam_re, lam_im);
    scan_partial_kernel<<<BATCH * NHEADS * NCHUNK, 64, 0, stream>>>(
        proj, u, dtv, lam_re, lam_im, sP_re, sP_im, pP_re, pP_im);
    scan_combine_kernel<<<(BATCH * NHEADS * DHEAD * DSTATE) / 256, 256, 0, stream>>>(
        sP_re, sP_im, pP_re, pP_im);
    scan_final_kernel<<<BATCH * NHEADS * NCHUNK, 64, 0, stream>>>(
        proj, u, dtv, lam_re, lam_im, sP_re, sP_im, d_skip, y);
  } else {
    scan_seq_kernel<<<BATCH * NHEADS, 64, 0, stream>>>(proj, u, dtv, gamma, theta,
                                                       d_skip, y);
  }
  // 4) Gate + RMS norm -> packed bf16 z
  gate_norm_kernel<<<BT, 256, 0, stream>>>(proj, nw, y, zbf);
  // 5) Output projection GEMM: out = z @ W_out^T   (4096 x 512, K=1024)
  {
    const int Ntiles = DMODEL / 16, ngroups = (Ntiles + 3) / 4;
    const int waves = (BT / 32) * ngroups;
    wmma_gemm_bf16_kernel<<<(waves + 7) / 8, 256, 0, stream>>>(zbf, wobf, out,
                                                               BT, DMODEL, DINNER);
  }
}
